// ASH_DGT_50319836840356
// MI455X (gfx1250) — 8-bit matrix operand order repaired and checked by reading, compiled, not run on hardware
//
#include <hip/hip_runtime.h>
#include <stdint.h>


#define N       4096
#define D       256
#define NSUP    6
#define MPAIRS  8386560u
#define NBLK    512
#define GEMM_BM 128
#define GEMM_BN 64
#define GEMM_BK 128

typedef int   v16i __attribute__((ext_vector_type(16)));
typedef float v8f  __attribute__((ext_vector_type(8)));

__global__ void k_pack_fp8(const float* __restrict__ A, unsigned char* __restrict__ A8) {
    for (size_t i = (size_t)blockIdx.x * blockDim.x + threadIdx.x;
         i < (size_t)N * N; i += (size_t)gridDim.x * blockDim.x)
        A8[i] = (A[i] > 0.0f) ? (unsigned char)0x38 : (unsigned char)0x00;
}

__global__ void k_deg(const float* __restrict__ A, float* __restrict__ deg) {
    __shared__ float red[256];
    const int row = blockIdx.x;
    float s = 0.f;
    for (int j = threadIdx.x; j < N; j += 256)
        s += (A[(size_t)row * N + j] > 0.0f) ? 1.0f : 0.0f;
    red[threadIdx.x] = s; __syncthreads();
    for (int off = 128; off > 0; off >>= 1) {
        if (threadIdx.x < off) red[threadIdx.x] += red[threadIdx.x + off];
        __syncthreads();
    }
    if (threadIdx.x == 0) deg[row] = red[0];
}

__global__ void __launch_bounds__(256)
k_jaccard_wmma(const unsigned char* __restrict__ A8,
               const float* __restrict__ deg,
               unsigned long long* __restrict__ keys) {
    __shared__ unsigned char lA[GEMM_BM * GEMM_BK];
    __shared__ unsigned char lB[GEMM_BN * GEMM_BK];
    const int i0 = blockIdx.y * GEMM_BM;
    const int j0 = blockIdx.x * GEMM_BN;
    if (i0 >= j0 + GEMM_BN) return;
    const int tid  = threadIdx.x;
    const int lane = tid & 31;
    const int wave = tid >> 5;
    const int wr = wave & 3;
    const int wc = wave >> 2;
    const int m  = lane & 15;
    const int hi = lane >> 4;

    const int rowA = tid >> 3, colA = (tid & 7) << 4;
    const int rowB = tid >> 3, colB = (tid & 7) << 4;

    v8f acc[2][2] = {};
    for (int kc = 0; kc < N; kc += GEMM_BK) {
        float4 tA[4], tB[2];
        #pragma unroll
        for (int r = 0; r < 4; ++r)
            tA[r] = *(const float4*)&A8[(size_t)(i0 + rowA + r * 32) * N + kc + colA];
        #pragma unroll
        for (int r = 0; r < 2; ++r)
            tB[r] = *(const float4*)&A8[(size_t)(j0 + rowB + r * 32) * N + kc + colB];
        #pragma unroll
        for (int r = 0; r < 4; ++r)
            *(float4*)&lA[(rowA + r * 32) * GEMM_BK + colA] = tA[r];
        #pragma unroll
        for (int r = 0; r < 2; ++r)
            *(float4*)&lB[(rowB + r * 32) * GEMM_BK + colB] = tB[r];

        if (kc + GEMM_BK < N) {
            __builtin_prefetch(&A8[(size_t)(i0 + (tid >> 1)) * N + kc + GEMM_BK], 0, 1);
            __builtin_prefetch(&A8[(size_t)(j0 + (tid >> 2)) * N + kc + GEMM_BK], 0, 1);
        }
        __syncthreads();

        v16i a[2], b[2];
        #pragma unroll
        for (int v = 0; v < 16; ++v) {
            const int v8 = v & 7;
            const int ka = ((v & 8) ? 64 : 0) + ((v8 >> 1) << 4) + (hi << 3) + ((v8 & 1) << 2);
            const int kb = ka;
            #pragma unroll
            for (int t = 0; t < 2; ++t) {
                a[t][v] = *(const int*)&lA[((wr * 2 + t) * 16 + m) * GEMM_BK + ka];
                b[t][v] = *(const int*)&lB[((wc * 2 + t) * 16 + m) * GEMM_BK + kb];
            }
        }
        #pragma unroll
        for (int ti = 0; ti < 2; ++ti)
            #pragma unroll
            for (int tj = 0; tj < 2; ++tj)
                acc[ti][tj] = __builtin_amdgcn_wmma_f32_16x16x128_fp8_fp8(
                    a[ti], b[tj], (short)0, acc[ti][tj], false, false);
        __syncthreads();
    }

    #pragma unroll
    for (int ti = 0; ti < 2; ++ti) {
        #pragma unroll
        for (int tj = 0; tj < 2; ++tj) {
            #pragma unroll
            for (int r = 0; r < 8; ++r) {
                const int gi = i0 + (wr * 2 + ti) * 16 + hi * 8 + r;
                const int gj = j0 + (wc * 2 + tj) * 16 + m;
                if (gi < gj) {
                    const float inter = acc[ti][tj][r];
                    const float uni   = deg[gi] + deg[gj] - inter;
                    const float sim   = (uni > 0.0f) ? (inter / fmaxf(uni, 1.0f)) : 0.0f;
                    const unsigned hb = ~__float_as_uint(sim);
                    const unsigned lo = ((unsigned)gi << 16) | (unsigned)gj;
                    const unsigned long long rank =
                        (unsigned long long)gi * (2ull * N - 1ull - (unsigned long long)gi) / 2ull
                        + (unsigned long long)(gj - gi - 1);
                    keys[rank] = ((unsigned long long)hb << 32) | (unsigned long long)lo;
                }
            }
        }
    }
}

__global__ void k_hist(const unsigned long long* __restrict__ keys,
                       unsigned* __restrict__ hist, int shift) {
    __shared__ unsigned h[256];
    h[threadIdx.x] = 0; __syncthreads();
    for (unsigned i = blockIdx.x * 256u + threadIdx.x; i < MPAIRS; i += NBLK * 256u)
        atomicAdd(&h[(unsigned)(keys[i] >> shift) & 255u], 1u);
    __syncthreads();
    hist[(unsigned)threadIdx.x * NBLK + blockIdx.x] = h[threadIdx.x];
}

__global__ void k_scan(unsigned* __restrict__ hist) {
    __shared__ unsigned buf[256];
    __shared__ unsigned carry;
    if (threadIdx.x == 0) carry = 0;
    __syncthreads();
    const int total = 256 * NBLK;
    for (int base = 0; base < total; base += 256) {
        const unsigned v = hist[base + threadIdx.x];
        buf[threadIdx.x] = v; __syncthreads();
        for (int off = 1; off < 256; off <<= 1) {
            const unsigned t = (threadIdx.x >= off) ? buf[threadIdx.x - off] : 0u;
            __syncthreads();
            buf[threadIdx.x] += t;
            __syncthreads();
        }
        const unsigned incl = buf[threadIdx.x];
        hist[base + threadIdx.x] = carry + incl - v;
        __syncthreads();
        if (threadIdx.x == 255) carry += incl;
        __syncthreads();
    }
}

__global__ void k_scatter(const unsigned long long* __restrict__ src,
                          unsigned long long* __restrict__ dst,
                          const unsigned* __restrict__ hist, int shift) {
    __shared__ unsigned off_[256];
    off_[threadIdx.x] = hist[(unsigned)threadIdx.x * NBLK + blockIdx.x];
    __syncthreads();
    for (unsigned i = blockIdx.x * 256u + threadIdx.x; i < MPAIRS; i += NBLK * 256u) {
        const unsigned long long k = src[i];
        const unsigned d = (unsigned)(k >> shift) & 255u;
        const unsigned pos = atomicAdd(&off_[d], 1u);
        dst[pos] = k;
    }
}

__global__ void k_unionfind(const unsigned long long* __restrict__ keys,
                            int* __restrict__ parent, int* __restrict__ rootlab,
                            int* __restrict__ labels, int* __restrict__ sizes,
                            float* __restrict__ scale) {
    if (threadIdx.x != 0 || blockIdx.x != 0) return;
    for (int i = 0; i < N; ++i) { parent[i] = i; rootlab[i] = -1; sizes[i] = 0; }
    int merged = 0;
    for (unsigned idx = 0; idx < MPAIRS && merged < N - NSUP; ++idx) {
        const unsigned lo = (unsigned)keys[idx];
        int pi = (int)(lo >> 16), pj = (int)(lo & 0xffffu);
        while (parent[pi] != pi) { parent[pi] = parent[parent[pi]]; pi = parent[pi]; }
        while (parent[pj] != pj) { parent[pj] = parent[parent[pj]]; pj = parent[pj]; }
        if (pi != pj) { parent[pj] = pi; ++merged; }
    }
    int nl = 0;
    for (int i = 0; i < N; ++i) {
        int r = i;
        while (parent[r] != r) { parent[r] = parent[parent[r]]; r = parent[r]; }
        if (rootlab[r] < 0) rootlab[r] = nl++;
        labels[i] = rootlab[r];
    }
    for (int i = 0; i < N; ++i) sizes[labels[i]]++;
    for (int c = 0; c < nl; ++c) scale[c] = rsqrtf((float)sizes[c] + 1e-10f);
}

__global__ void k_zero(float* __restrict__ out, int n) {
    for (int i = blockIdx.x * blockDim.x + threadIdx.x; i < n; i += gridDim.x * blockDim.x)
        out[i] = 0.0f;
}

__global__ void k_P(const int* __restrict__ labels, const float* __restrict__ scale,
                    float* __restrict__ P) {
    const int i = blockIdx.x * blockDim.x + threadIdx.x;
    if (i < N) { const int l = labels[i]; if (l < NSUP) P[i * NSUP + l] = scale[l]; }
}

__global__ void k_xcoarse(const float* __restrict__ X, const int* __restrict__ labels,
                          const float* __restrict__ scale, float* __restrict__ out) {
    const int m = blockIdx.x, d = threadIdx.x;
    float s = 0.f;
    for (int i = 0; i < N; ++i)
        if (labels[i] == m) s += X[(size_t)i * D + d];
    out[m * D + d] = s * scale[m];
}

__global__ void k_AP(const float* __restrict__ A, const int* __restrict__ labels,
                     const float* __restrict__ scale, float* __restrict__ Y) {
    __shared__ float red[256];
    const int i = blockIdx.x;
    float acc[NSUP] = {0, 0, 0, 0, 0, 0};
    for (int j = threadIdx.x; j < N; j += 256) {
        const float a = A[(size_t)i * N + j];
        if (a != 0.0f) { const int l = labels[j]; if (l < NSUP) acc[l] += a * scale[l]; }
    }
    for (int c = 0; c < NSUP; ++c) {
        red[threadIdx.x] = acc[c]; __syncthreads();
        for (int off = 128; off > 0; off >>= 1) {
            if (threadIdx.x < off) red[threadIdx.x] += red[threadIdx.x + off];
            __syncthreads();
        }
        if (threadIdx.x == 0) Y[i * NSUP + c] = red[0];
        __syncthreads();
    }
}

__global__ void k_acoarse(const float* __restrict__ Y, const int* __restrict__ labels,
                          const float* __restrict__ scale, float* __restrict__ out) {
    __shared__ float red[256];
    const int m1 = blockIdx.x / NSUP, m2 = blockIdx.x % NSUP;
    float s = 0.f;
    for (int i = threadIdx.x; i < N; i += 256)
        if (labels[i] == m1) s += Y[i * NSUP + m2];
    red[threadIdx.x] = s; __syncthreads();
    for (int off = 128; off > 0; off >>= 1) {
        if (threadIdx.x < off) red[threadIdx.x] += red[threadIdx.x + off];
        __syncthreads();
    }
    if (threadIdx.x == 0) out[m1 * NSUP + m2] = red[0] * scale[m1];
}

extern "C" void kernel_launch(void* const* d_in, const int* in_sizes, int n_in,
                              void* d_out, int out_size, void* d_ws, size_t ws_size,
                              hipStream_t stream) {
    (void)in_sizes; (void)n_in; (void)ws_size;
    const float* X = (const float*)d_in[0];
    const float* A = (const float*)d_in[1];
    float* out = (float*)d_out;

    char* ws = (char*)d_ws;
    size_t o = 0;
    auto alloc = [&](size_t b) { size_t r = o; o = (o + b + 255) & ~(size_t)255; return r; };
    unsigned char*      A8      = (unsigned char*)     (ws + alloc((size_t)N * N));
    float*              deg     = (float*)             (ws + alloc((size_t)N * 4));
    unsigned long long* keys0   = (unsigned long long*)(ws + alloc((size_t)MPAIRS * 8));
    unsigned long long* keys1   = (unsigned long long*)(ws + alloc((size_t)MPAIRS * 8));
    unsigned*           hist    = (unsigned*)          (ws + alloc((size_t)256 * NBLK * 4));
    int*                parent  = (int*)               (ws + alloc((size_t)N * 4));
    int*                rootlab = (int*)               (ws + alloc((size_t)N * 4));
    int*                labels  = (int*)               (ws + alloc((size_t)N * 4));
    int*                sizes   = (int*)               (ws + alloc((size_t)N * 4));
    float*              scale   = (float*)             (ws + alloc((size_t)N * 4));
    float*              Y       = (float*)             (ws + alloc((size_t)N * NSUP * 4));

    float* outX = out;
    float* outA = out + NSUP * D;
    float* outP = outA + NSUP * NSUP;

    k_pack_fp8<<<2048, 256, 0, stream>>>(A, A8);
    k_deg<<<N, 256, 0, stream>>>(A, deg);
    k_jaccard_wmma<<<dim3(N / GEMM_BN, N / GEMM_BM), 256, 0, stream>>>(A8, deg, keys0);

    unsigned long long* src = keys0;
    unsigned long long* dst = keys1;
    for (int p = 0; p < 8; ++p) {
        k_hist<<<NBLK, 256, 0, stream>>>(src, hist, p * 8);
        k_scan<<<1, 256, 0, stream>>>(hist);
        k_scatter<<<NBLK, 256, 0, stream>>>(src, dst, hist, p * 8);
        unsigned long long* t = src; src = dst; dst = t;
    }

    k_zero<<<128, 256, 0, stream>>>(out, out_size);
    k_unionfind<<<1, 1, 0, stream>>>(src, parent, rootlab, labels, sizes, scale);
    k_P<<<(N + 255) / 256, 256, 0, stream>>>(labels, scale, outP);
    k_xcoarse<<<NSUP, 256, 0, stream>>>(X, labels, scale, outX);
    k_AP<<<N, 256, 0, stream>>>(A, labels, scale, Y);
    k_acoarse<<<NSUP * NSUP, 256, 0, stream>>>(Y, labels, scale, outA);
}
